// ContextAwareYieldGNN_11570641895684
// MI455X (gfx1250) — matrix operand order repaired and checked by reading, not compiled, not run on hardware
//
#include <hip/hip_runtime.h>
#include <hip/hip_fp16.h>

#define N_NODES   100000
#define N_EDGES   1600000
#define ETOT      1700000      // edges + self loops
#define HEADS     4
#define CH        64
#define HC        256          // HEADS*CH
#define NEG_SLOPE 0.2f
#define BN_EPS    1e-5f

typedef __attribute__((ext_vector_type(16))) _Float16 v16h;
typedef __attribute__((ext_vector_type(8)))  _Float16 v8h;
typedef __attribute__((ext_vector_type(8)))  float    v8f;
typedef __attribute__((ext_vector_type(4)))  float    v4f;

// ---------- monotone float <-> uint encoding for atomicMax on signed floats ----------
__device__ __forceinline__ unsigned enc_f(float f) {
  unsigned u = __float_as_uint(f);
  return (u & 0x80000000u) ? ~u : (u | 0x80000000u);
}
__device__ __forceinline__ float dec_f(unsigned u) {
  return __uint_as_float((u & 0x80000000u) ? (u ^ 0x80000000u) : ~u);
}

// ---------- cast fp32 -> fp16 ----------
__global__ void cast_f32_to_f16(const float* __restrict__ src, _Float16* __restrict__ dst, int n) {
  int i = blockIdx.x * blockDim.x + threadIdx.x;
  if (i < n) dst[i] = (_Float16)src[i];
}

// ---------- pack W [Fin,256] fp32 row-major into WMMA B-operand lane layout (f16) ----------
// Wp[(((kt*16+nt)*32)+lane)*16 + e] = W[(kt*32 + 16*(lane/16) + e)*256 + nt*16 + (lane&15)]
__global__ void pack_w_kernel(const float* __restrict__ W, _Float16* __restrict__ Wp, int Fin) {
  int tid = blockIdx.x * blockDim.x + threadIdx.x;
  if (tid >= Fin * HC) return;
  int e    = tid & 15;
  int lane = (tid >> 4) & 31;
  int rest = tid >> 9;          // kt*16 + nt
  int nt   = rest & 15;
  int kt   = rest >> 4;
  int k = kt * 32 + 16 * (lane >> 4) + e;
  int n = nt * 16 + (lane & 15);
  Wp[tid] = (_Float16)W[(size_t)k * HC + n];
}

// ---------- WMMA GEMM: Hout[N,256] = A[N,Fin] @ W[Fin,256], f16 in / f32 out ----------
// one wave per 16x16 output tile; tiles = 6250(M) * 16(N) = 100000; 8 waves/block
__global__ void __launch_bounds__(256) gemm_wmma_f16(const _Float16* __restrict__ A,
                                                     const _Float16* __restrict__ Wp,
                                                     float* __restrict__ Hout, int Fin) {
  const int lane = threadIdx.x & 31;
  const int wave = threadIdx.x >> 5;
  const int tile = blockIdx.x * 8 + wave;   // 0..99999 (exact)
  const int tm   = tile >> 4;               // 0..6249
  const int nt   = tile & 15;               // 0..15
  const int g    = lane >> 4;
  const int mn   = lane & 15;
  const int row  = tm * 16 + mn;
  const int Kt   = Fin >> 5;

  v8f c = {};
  for (int kt = 0; kt < Kt; ++kt) {
    const _Float16* abase = A + (size_t)row * Fin + kt * 32;
    v8h alo = *(const v8h*)(abase + 16 * g);       // K = 16g .. 16g+7
    v8h ahi = *(const v8h*)(abase + 16 * g + 8);   // K = 16g+8 .. 16g+15
    v16h a;
#pragma unroll
    for (int i = 0; i < 8; ++i) { a[i] = alo[i]; a[i + 8] = ahi[i]; }
    v16h b = *(const v16h*)(Wp + (size_t)(((kt * 16 + nt) * 32) + lane) * 16);
    c = __builtin_amdgcn_wmma_f32_16x16x32_f16(false, a, false, b, (short)0, c, false, false);
  }
  const int col = nt * 16 + mn;
  float* out = Hout + (size_t)(tm * 16) * HC + col;
#pragma unroll
  for (int r = 0; r < 8; ++r) out[(size_t)(r + 8 * g) * HC] = c[r];
}

// ---------- per-node attention logits: e_src[n,h] = sum_c H[n,h,c]*a_src[h,c] ----------
// one wave per node; lane covers 8 consecutive channels of flattened [H*C]
__global__ void __launch_bounds__(256) alpha_kernel(const float* __restrict__ Hf,
                                                    const float* __restrict__ a_src,
                                                    const float* __restrict__ a_dst,
                                                    float* __restrict__ esrc,
                                                    float* __restrict__ edst) {
  const int lane = threadIdx.x & 31;
  const int node = blockIdx.x * 8 + (threadIdx.x >> 5);   // exact: 12500*8 = 100000
  const v4f* hv = (const v4f*)(Hf + (size_t)node * HC + lane * 8);
  v4f h0 = hv[0], h1 = hv[1];
  const v4f* sv = (const v4f*)(a_src + lane * 8);
  const v4f* dv = (const v4f*)(a_dst + lane * 8);
  v4f s0 = sv[0], s1 = sv[1], d0 = dv[0], d1 = dv[1];
  float ps = 0.f, pd = 0.f;
#pragma unroll
  for (int i = 0; i < 4; ++i) {
    ps += h0[i] * s0[i] + h1[i] * s1[i];
    pd += h0[i] * d0[i] + h1[i] * d1[i];
  }
  for (int off = 4; off; off >>= 1) {
    ps += __shfl_down(ps, off, 32);
    pd += __shfl_down(pd, off, 32);
  }
  if ((lane & 7) == 0) {
    int head = lane >> 3;
    esrc[node * 4 + head] = ps;
    edst[node * 4 + head] = pd;
  }
}

// ---------- edge pass 1: leaky-relu logits + segment max via integer atomicMax ----------
__global__ void edge_max_kernel(const int* __restrict__ ei,
                                const float* __restrict__ esrc, const float* __restrict__ edst,
                                float* __restrict__ eedge, unsigned* __restrict__ menc) {
  int e = blockIdx.x * blockDim.x + threadIdx.x;
  if (e >= ETOT) return;
  int s, d;
  if (e < N_EDGES) { s = ei[e]; d = ei[N_EDGES + e]; }
  else             { s = d = e - N_EDGES; }            // self loop
  v4f es = *(const v4f*)(esrc + (size_t)s * 4);
  v4f ed = *(const v4f*)(edst + (size_t)d * 4);
  v4f out;
#pragma unroll
  for (int h = 0; h < 4; ++h) {
    float v = es[h] + ed[h];
    v = v > 0.f ? v : NEG_SLOPE * v;
    out[h] = v;
    atomicMax(&menc[(size_t)d * 4 + h], enc_f(v));
  }
  *(v4f*)(eedge + (size_t)e * 4) = out;
}

// ---------- edge pass 2: w = exp(e - max); denom += w; acc[dst] += w * H[src] ----------
// one wave per edge; lane covers 8 channels of flattened [H*C]
__global__ void __launch_bounds__(256) edge_scatter_kernel(const int* __restrict__ ei,
                                                           const float* __restrict__ eedge,
                                                           const unsigned* __restrict__ menc,
                                                           const float* __restrict__ Hf,
                                                           float* __restrict__ acc,
                                                           float* __restrict__ denom) {
  const int lane = threadIdx.x & 31;
  const int e = blockIdx.x * 8 + (threadIdx.x >> 5);    // exact: 212500*8 = 1700000
  int s, d;
  if (e < N_EDGES) { s = ei[e]; d = ei[N_EDGES + e]; }
  else             { s = d = e - N_EDGES; }
  const int head = lane >> 3;
  float w = __expf(eedge[(size_t)e * 4 + head] - dec_f(menc[(size_t)d * 4 + head]));
  const v4f* hs = (const v4f*)(Hf + (size_t)s * HC + lane * 8);
  v4f h0 = hs[0], h1 = hs[1];
  float* ap = acc + (size_t)d * HC + lane * 8;
#pragma unroll
  for (int i = 0; i < 4; ++i) {
    unsafeAtomicAdd(ap + i,     h0[i] * w);
    unsafeAtomicAdd(ap + 4 + i, h1[i] * w);
  }
  if (lane < 4) {
    float wd = __expf(eedge[(size_t)e * 4 + lane] - dec_f(menc[(size_t)d * 4 + lane]));
    unsafeAtomicAdd(&denom[(size_t)d * 4 + lane], wd);
  }
}

// ---------- finalize: head-mean + bias + BN + ReLU -> next layer input [N,64] ----------
__global__ void finalize_kernel(const float* __restrict__ acc, const float* __restrict__ denom,
                                const float* __restrict__ bias, const float* __restrict__ gamma,
                                const float* __restrict__ beta, const float* __restrict__ mean,
                                const float* __restrict__ var, float* __restrict__ xout) {
  int tid = blockIdx.x * blockDim.x + threadIdx.x;     // exact: N*64
  int n = tid >> 6, c = tid & 63;
  const float* arow = acc + (size_t)n * HC + c;
  const float* dr = denom + (size_t)n * 4;
  float s = 0.f;
#pragma unroll
  for (int h = 0; h < 4; ++h) s += arow[h * 64] / dr[h];
  float v = s * 0.25f + bias[c];
  v = (v - mean[c]) * (gamma[c] * rsqrtf(var[c] + BN_EPS)) + beta[c];
  xout[tid] = v > 0.f ? v : 0.f;
}

// ---------- MLP head: relu(x@W1+b1)@W2+b2 ----------
__global__ void __launch_bounds__(256) mlp_kernel(const float* __restrict__ x,
                                                  const float* __restrict__ w1, const float* __restrict__ b1,
                                                  const float* __restrict__ w2, const float* __restrict__ b2,
                                                  float* __restrict__ out) {
  __shared__ float sw1[64 * 32];
  __shared__ float sb1[32];
  __shared__ float sw2[32];
  for (int i = threadIdx.x; i < 64 * 32; i += 256) sw1[i] = w1[i];
  if (threadIdx.x < 32) { sb1[threadIdx.x] = b1[threadIdx.x]; sw2[threadIdx.x] = w2[threadIdx.x]; }
  __syncthreads();
  int n = blockIdx.x * blockDim.x + threadIdx.x;
  if (n >= N_NODES) return;
  float xr[64];
  const v4f* xv = (const v4f*)(x + (size_t)n * 64);
#pragma unroll
  for (int i = 0; i < 16; ++i) {
    v4f t = xv[i];
    xr[4*i] = t[0]; xr[4*i+1] = t[1]; xr[4*i+2] = t[2]; xr[4*i+3] = t[3];
  }
  float o = b2[0];
#pragma unroll 4
  for (int j = 0; j < 32; ++j) {
    float hj = sb1[j];
    for (int i = 0; i < 64; ++i) hj += xr[i] * sw1[i * 32 + j];
    hj = hj > 0.f ? hj : 0.f;
    o += hj * sw2[j];
  }
  out[n] = o;
}

extern "C" void kernel_launch(void* const* d_in, const int* in_sizes, int n_in,
                              void* d_out, int out_size, void* d_ws, size_t ws_size,
                              hipStream_t stream) {
  const float* x        = (const float*)d_in[0];
  const int*   ei       = (const int*)d_in[1];
  const float* W0       = (const float*)d_in[2];
  const float* Wl       = (const float*)d_in[3];
  const float* att_src  = (const float*)d_in[4];
  const float* att_dst  = (const float*)d_in[5];
  const float* conv_b   = (const float*)d_in[6];
  const float* bn_gamma = (const float*)d_in[7];
  const float* bn_beta  = (const float*)d_in[8];
  const float* bn_mean  = (const float*)d_in[9];
  const float* bn_var   = (const float*)d_in[10];
  const float* lin1_w   = (const float*)d_in[11];
  const float* lin1_b   = (const float*)d_in[12];
  const float* lin2_w   = (const float*)d_in[13];
  const float* lin2_b   = (const float*)d_in[14];

  // workspace carve-out (256B aligned); total ~289 MB
  char* ws = (char*)d_ws;
  size_t off = 0;
  auto carve = [&](size_t bytes) -> char* {
    char* p = ws + off;
    off = (off + bytes + 255) & ~(size_t)255;
    return p;
  };
  float*    Hf    = (float*)   carve((size_t)N_NODES * HC * 4);   // GEMM output [N,256]
  float*    acc   = (float*)   carve((size_t)N_NODES * HC * 4);   // message accumulator
  _Float16* xh    = (_Float16*)carve((size_t)N_NODES * 128 * 2);  // f16 activations (max Fin)
  float*    xcur  = (float*)   carve((size_t)N_NODES * CH * 4);   // layer output [N,64]
  _Float16* Wp    = (_Float16*)carve((size_t)128 * HC * 2);       // packed f16 weights
  float*    esrc  = (float*)   carve((size_t)N_NODES * 4 * 4);
  float*    edst  = (float*)   carve((size_t)N_NODES * 4 * 4);
  float*    eedge = (float*)   carve((size_t)ETOT * 4 * 4);
  unsigned* menc  = (unsigned*)carve((size_t)N_NODES * 4 * 4);
  float*    denom = (float*)   carve((size_t)N_NODES * 4 * 4);

  for (int l = 0; l < 3; ++l) {
    const int Fin = (l == 0) ? 128 : 64;
    const float* xin = (l == 0) ? x : xcur;
    const float* W   = (l == 0) ? W0 : (Wl + (size_t)(l - 1) * 64 * HC);

    cast_f32_to_f16<<<(N_NODES * Fin) / 256, 256, 0, stream>>>(xin, xh, N_NODES * Fin);
    pack_w_kernel<<<Fin, 256, 0, stream>>>(W, Wp, Fin);

    hipMemsetAsync(menc,  0, (size_t)N_NODES * 4 * 4, stream);   // 0 == enc(below -inf)
    hipMemsetAsync(denom, 0, (size_t)N_NODES * 4 * 4, stream);
    hipMemsetAsync(acc,   0, (size_t)N_NODES * HC * 4, stream);

    gemm_wmma_f16<<<12500, 256, 0, stream>>>(xh, Wp, Hf, Fin);   // 100000 tiles / 8 waves
    alpha_kernel<<<12500, 256, 0, stream>>>(Hf, att_src + l * HC, att_dst + l * HC, esrc, edst);
    edge_max_kernel<<<(ETOT + 255) / 256, 256, 0, stream>>>(ei, esrc, edst, eedge, menc);
    edge_scatter_kernel<<<ETOT / 8, 256, 0, stream>>>(ei, eedge, menc, Hf, acc, denom);
    finalize_kernel<<<(N_NODES * CH) / 256, 256, 0, stream>>>(
        acc, denom, conv_b + l * CH, bn_gamma + l * CH, bn_beta + l * CH,
        bn_mean + l * CH, bn_var + l * CH, xcur);
  }

  mlp_kernel<<<(N_NODES + 255) / 256, 256, 0, stream>>>(
      xcur, lin1_w, lin1_b, lin2_w, lin2_b, (float*)d_out);
}
